// StridedBlockAttention_61426622267610
// MI455X (gfx1250) — matrix operand order repaired and checked by reading, not compiled, not run on hardware
//
#include <hip/hip_runtime.h>
#include <hip/hip_bf16.h>

typedef __attribute__((ext_vector_type(16))) __bf16 v16bf;
typedef __attribute__((ext_vector_type(8)))  float  v8f;

#define D_MODEL 512
#define NHEAD   8
#define DK      64
#define NBLK    10
#define NSEL    80          // on-stride rows total (8 per block x 10 blocks)
#define SEQ     4000
// fp32 value of mean over a masked column: (-1e9 * 3992)/4000, real scores absorbed
#define G_ON    (-998000000.0f)

__device__ __forceinline__ int sel_pos(int r) {   // r in [0,80) -> global seq position
    return (r >> 3) * 400 + (r & 7) * 50;
}

// ---------------------------------------------------------------------------
// Kernel 1: gathered QKV projections.  dst[80x512] = gather(src)[80x512] @ W^T + b
// One wave (32 threads) per 16x16 output tile; K-loop of 32 with bf16 WMMA.
// grid = (5 mtiles, 32 ntiles, 3 matrices)
// ---------------------------------------------------------------------------
__global__ __launch_bounds__(32)
void qkv_wmma(const float* __restrict__ q, const float* __restrict__ k,
              const float* __restrict__ v,
              const float* __restrict__ Wq, const float* __restrict__ bq,
              const float* __restrict__ Wk, const float* __restrict__ bk,
              const float* __restrict__ Wv, const float* __restrict__ bv,
              float* __restrict__ oq, float* __restrict__ ok, float* __restrict__ ov)
{
    const int mtile = blockIdx.x;          // 0..4  -> 80 gathered rows
    const int ntile = blockIdx.y;          // 0..31 -> 512 output features
    const int z     = blockIdx.z;          // 0:Q 1:K 2:V
    const float* src  = (z == 0) ? q  : (z == 1) ? k  : v;
    const float* W    = (z == 0) ? Wq : (z == 1) ? Wk : Wv;
    const float* bias = (z == 0) ? bq : (z == 1) ? bk : bv;
    float*       dst  = (z == 0) ? oq : (z == 1) ? ok : ov;

    const int lane = threadIdx.x & 31;
    const int half = (lane >> 4) & 1;      // which lane-half
    const int l15  = lane & 15;

    const int arow = sel_pos(mtile * 16 + l15);   // gathered sequence row (M = l15)
    const int ncol = ntile * 16 + l15;            // output feature (N = l15), W row

    // A (16x32 bf16): lane-half 0 holds K {0..7,16..23}, half 1 holds K {8..15,24..31}
    const float* aptr = src + arow * D_MODEL + half * 8;
    // B (32x16 bf16): lane-half 0 holds K {0..7,16..23}, half 1 holds K {8..15,24..31} of column ncol
    const float* bptr = W + ncol * D_MODEL + half * 8;

    v8f c = {};
    for (int kk = 0; kk < D_MODEL; kk += 32) {
        const float* ap = aptr + kk;
        const float* bp = bptr + kk;
        v16bf a, b;
        #pragma unroll
        for (int e = 0; e < 8; ++e) {
            a[e]     = (__bf16)ap[e];
            a[8 + e] = (__bf16)ap[16 + e];
        }
        #pragma unroll
        for (int e = 0; e < 8; ++e) {
            b[e]     = (__bf16)bp[e];
            b[8 + e] = (__bf16)bp[16 + e];
        }
        c = __builtin_amdgcn_wmma_f32_16x16x32_bf16(false, a, false, b,
                                                    (short)0, c, false, false);
    }
    const float bn = bias[ncol];
    #pragma unroll
    for (int g = 0; g < 8; ++g) {          // C/D: VGPR g -> row g (half 0) / 8+g (half 1)
        const int row = mtile * 16 + half * 8 + g;
        dst[row * D_MODEL + ncol] = c[g] + bn;
    }
}

// ---------------------------------------------------------------------------
// Kernel 2: per (head, block) 8x8 attention.  Literally emulates the reference's
// fp32 `score + global_mean` (the +G_ON absorbs the score exactly like the ref).
// Writes context rows into xs[0..79] and per-block V sums.
// grid = 80 (h*10+b), 64 threads.
// ---------------------------------------------------------------------------
__global__ __launch_bounds__(64)
void attn_ctx(const float* __restrict__ Qs, const float* __restrict__ Ks,
              const float* __restrict__ Vs, float* __restrict__ xs,
              float* __restrict__ Vblk)
{
    __shared__ float q8[8][DK], k8[8][DK], v8[8][DK];
    __shared__ float sc[8][8], p[8][8];

    const int w = blockIdx.x;
    const int h = w / NBLK, b = w % NBLK;
    const int t = threadIdx.x;             // 0..63

    #pragma unroll
    for (int i = 0; i < 8; ++i) {
        const int off = (b * 8 + i) * D_MODEL + h * DK + t;
        q8[i][t] = Qs[off];
        k8[i][t] = Ks[off];
        v8[i][t] = Vs[off];
    }
    __syncthreads();

    // scores: 64 threads = 64 (i,j) pairs
    {
        const int i = t >> 3, j = t & 7;
        float acc = 0.f;
        #pragma unroll
        for (int d = 0; d < DK; ++d) acc += q8[i][d] * k8[j][d];
        sc[i][j] = acc * 0.125f;           // / sqrt(64)
    }
    __syncthreads();

    // softmax over the 8 surviving keys; + G_ON reproduces the reference fp32 path
    {
        const int i = t >> 3, j = t & 7;
        float row[8], m = -3.4e38f;
        #pragma unroll
        for (int jj = 0; jj < 8; ++jj) {
            row[jj] = sc[i][jj] + G_ON;    // fp32 absorption, identical to reference
            m = fmaxf(m, row[jj]);
        }
        float Z = 0.f;
        #pragma unroll
        for (int jj = 0; jj < 8; ++jj) Z += expf(row[jj] - m);
        p[i][j] = expf(row[j] - m) / Z;
    }
    __syncthreads();

    // context: thread t = head-dim d
    float vb = 0.f;
    #pragma unroll
    for (int j = 0; j < 8; ++j) vb += v8[j][t];
    Vblk[(h * NBLK + b) * DK + t] = vb;

    #pragma unroll
    for (int i = 0; i < 8; ++i) {
        float acc = 0.f;
        #pragma unroll
        for (int j = 0; j < 8; ++j) acc += p[i][j] * v8[j][t];
        xs[(b * 8 + i) * D_MODEL + h * DK + t] = acc;
    }
}

// ---------------------------------------------------------------------------
// Kernel 3: row 80 of xs = broadcast context for all off-stride rows
//           (uniform 1/80 attention over the 80 on-stride keys), zero pad rows.
// ---------------------------------------------------------------------------
__global__ __launch_bounds__(512)
void build_xoff(const float* __restrict__ Vblk, float* __restrict__ xs)
{
    const int d = threadIdx.x;             // 0..511
    const int h = d >> 6, dd = d & 63;
    float s = 0.f;
    #pragma unroll
    for (int b = 0; b < NBLK; ++b) s += Vblk[(h * NBLK + b) * DK + dd];
    xs[NSEL * D_MODEL + d] = s * (1.0f / 80.0f);
    #pragma unroll
    for (int r = NSEL + 1; r < 96; ++r) xs[r * D_MODEL + d] = 0.f;
}

// ---------------------------------------------------------------------------
// Kernel 4: output projection for the 81 unique rows (padded to 96).
// ytmp[96x512] = xs[96x512] @ Wo^T + bo.  Same WMMA tiling as kernel 1.
// grid = (6 mtiles, 32 ntiles)
// ---------------------------------------------------------------------------
__global__ __launch_bounds__(32)
void oproj_wmma(const float* __restrict__ xs, const float* __restrict__ Wo,
                const float* __restrict__ bo, float* __restrict__ ytmp)
{
    const int mtile = blockIdx.x;
    const int ntile = blockIdx.y;
    const int lane = threadIdx.x & 31;
    const int half = (lane >> 4) & 1;
    const int l15  = lane & 15;

    const int ncol = ntile * 16 + l15;
    const float* aptr = xs + (mtile * 16 + l15) * D_MODEL + half * 8;
    const float* bptr = Wo + ncol * D_MODEL + half * 8;

    v8f c = {};
    for (int kk = 0; kk < D_MODEL; kk += 32) {
        const float* ap = aptr + kk;
        const float* bp = bptr + kk;
        v16bf a, b;
        #pragma unroll
        for (int e = 0; e < 8; ++e) {
            a[e]     = (__bf16)ap[e];
            a[8 + e] = (__bf16)ap[16 + e];
        }
        #pragma unroll
        for (int e = 0; e < 8; ++e) {
            b[e]     = (__bf16)bp[e];
            b[8 + e] = (__bf16)bp[16 + e];
        }
        c = __builtin_amdgcn_wmma_f32_16x16x32_bf16(false, a, false, b,
                                                    (short)0, c, false, false);
    }
    const float bn = bo[ncol];
    #pragma unroll
    for (int g = 0; g < 8; ++g) {
        const int row = mtile * 16 + half * 8 + g;
        ytmp[row * D_MODEL + ncol] = c[g] + bn;
    }
}

// ---------------------------------------------------------------------------
// Kernel 5: scatter/broadcast to the full [4000 x 512] output (the 8 MB
// HBM-bound part).  float4 stores; source rows live in L2 (81 x 2 KB).
// ---------------------------------------------------------------------------
__global__ __launch_bounds__(128)
void scatter_rows(const float* __restrict__ ytmp, float* __restrict__ out)
{
    const int i  = blockIdx.x;             // 0..3999
    const int t  = threadIdx.x;            // 0..127 (float4 lanes)
    const int ib = i % 400;
    const int r  = ((ib % 50) == 0) ? (i / 400) * 8 + ib / 50 : NSEL;
    const float4* src = (const float4*)(ytmp + r * D_MODEL);
    float4*       dst = (float4*)(out + i * D_MODEL);
    dst[t] = src[t];
}

// ---------------------------------------------------------------------------
extern "C" void kernel_launch(void* const* d_in, const int* in_sizes, int n_in,
                              void* d_out, int out_size, void* d_ws, size_t ws_size,
                              hipStream_t stream)
{
    const float* q  = (const float*)d_in[0];
    const float* k  = (const float*)d_in[1];
    const float* v  = (const float*)d_in[2];
    const float* Wq = (const float*)d_in[3];
    const float* bq = (const float*)d_in[4];
    const float* Wk = (const float*)d_in[5];
    const float* bk = (const float*)d_in[6];
    const float* Wv = (const float*)d_in[7];
    const float* bv = (const float*)d_in[8];
    const float* Wo = (const float*)d_in[9];
    const float* bo = (const float*)d_in[10];
    float* out = (float*)d_out;

    float* ws   = (float*)d_ws;
    float* Qs   = ws;                       // 80*512
    float* Ks   = Qs  + NSEL * D_MODEL;     // 80*512
    float* Vs   = Ks  + NSEL * D_MODEL;     // 80*512
    float* xs   = Vs  + NSEL * D_MODEL;     // 96*512 (81 used)
    float* Vbl  = xs  + 96 * D_MODEL;       // 8*10*64
    float* ytmp = Vbl + NHEAD * NBLK * DK;  // 96*512   (~905 KB total)

    qkv_wmma  <<<dim3(5, 32, 3), 32, 0, stream>>>(q, k, v, Wq, bq, Wk, bk, Wv, bv,
                                                  Qs, Ks, Vs);
    attn_ctx  <<<NHEAD * NBLK, 64, 0, stream>>>(Qs, Ks, Vs, xs, Vbl);
    build_xoff<<<1, 512, 0, stream>>>(Vbl, xs);
    oproj_wmma<<<dim3(6, 32), 32, 0, stream>>>(xs, Wo, bo, ytmp);
    scatter_rows<<<SEQ, 128, 0, stream>>>(ytmp, out);
}
